// LatentNeuralSDE_24996709662818
// MI455X (gfx1250) — compile-verified, matrix operand order checked by reading, not run on hardware
//
#include <hip/hip_runtime.h>
#include <cstdint>
#include <cstddef>

// Problem sizes (match reference)
#define OBS    512
#define LAT    64
#define BATCH  256
#define TSTEPS 200
#define ENC_H  1024
#define DEC_H  1024
#define SDE_H  256

typedef _Float16 half8   __attribute__((ext_vector_type(8)));
typedef _Float16 half16  __attribute__((ext_vector_type(16)));
typedef float    floatx8 __attribute__((ext_vector_type(8)));
typedef unsigned int uint32x4 __attribute__((ext_vector_type(4)));
typedef int      int32x8 __attribute__((ext_vector_type(8)));
typedef int      int32x4 __attribute__((ext_vector_type(4)));

// Padded LDS row strides (halves). Multiples of 8 halves (16B, keeps
// ds_load_b128 aligned) but NOT powers of two: row r starts at DWORD
// 36*r (mod 64) resp. 132*r (mod 64) -> 16 lanes hit 16 distinct bank
// quads -> conflict-free fragment loads.
#define ST64   72     // for K=64 matrices   (144B rows)
#define ST256  264    // for K=256 matrices  (528B rows)

// ---------------------------------------------------------------------------
// WMMA helpers (CDNA5 wave32, V_WMMA_F32_16X16X32_F16)
// 16-bit A-matrix 16x32 layout (ISA 7.12.2): lane L holds row m = L&15;
// lanes 0-15 carry K runs [0..7],[16..23]; lanes 16-31 carry [8..15],[24..31].
// Activations row-major [M][K], weights transposed [N][K] -> both fragments
// are two contiguous 8-half (16B) loads per lane.
// ---------------------------------------------------------------------------
static __device__ __forceinline__ half16 load_frag(const _Float16* p, int ld) {
  const int lane = threadIdx.x & 31;
  const int r    = lane & 15;
  const int hi   = lane >> 4;
  const _Float16* q = p + (size_t)r * ld + hi * 8;
  half8 a = *(const half8*)(q);
  half8 b = *(const half8*)(q + 16);
  return __builtin_shufflevector(a, b, 0,1,2,3,4,5,6,7,8,9,10,11,12,13,14,15);
}

static __device__ __forceinline__ floatx8 wmma_f16(half16 a, half16 b, floatx8 c) {
  return __builtin_amdgcn_wmma_f32_16x16x32_f16(false, a, false, b, (short)0, c,
                                                false, false);
}

static __device__ __forceinline__ floatx8 f8zero() {
  floatx8 z = {0.f, 0.f, 0.f, 0.f, 0.f, 0.f, 0.f, 0.f};
  return z;
}

// ---------------------------------------------------------------------------
// Tensor Data Mover: 2-D tile load, global [512 rows x 64 halves, row stride
// 1024 halves] -> LDS rows padded to 144B (pad 4 DWORDs after every 32 DWORDs)
// so WMMA fragment reads are bank-conflict-free.  D# per CDNA5 ISA:
//   group0: count=1 | lds_addr | global_addr | type=2
//   group1: data_size=2B, pad_enable, pad_interval=32dw, pad_amount=4dw,
//           tensor_dim0=1024, tensor_dim1=512, tile_dim0=64, tile_dim1=512,
//           tensor_dim0_stride=1024
// ---------------------------------------------------------------------------
static __device__ __forceinline__ void tdm_load_w2_chunk(unsigned lds_off,
                                                         const _Float16* gsrc) {
  const unsigned long long ga = (unsigned long long)(uintptr_t)gsrc;
  uint32x4 g0;
  g0[0] = 1u;                                        // count=1, user mode
  g0[1] = lds_off;                                   // LDS byte address
  g0[2] = (unsigned)ga;                              // global addr [31:0]
  g0[3] = ((unsigned)(ga >> 32) & 0x01FFFFFFu) | 0x80000000u; // [56:32]|type=2
  int32x8 g1;
  g1[0] = (int)(0x00010000u      // data_size = 1 -> 2 bytes/elem
              | 0x00100000u      // pad_enable
              | (4u << 22)       // pad_interval: 32 DWORDs (128B)
              | (3u << 25));     // pad_amount:   4 DWORDs  (16B)
  g1[1] = (int)(1024u << 16);    // tensor_dim0 = 1024 (bits 79:48, low part)
  g1[2] = (int)(512u << 16);     // tensor_dim1 = 512  (bits 111:80, low part)
  g1[3] = (int)(64u << 16);      // tile_dim0   = 64   (bits 127:112)
  g1[4] = 512;                   // tile_dim1   = 512  (bits 143:128)
  g1[5] = 1024;                  // tensor_dim0_stride = 1024 (bits 207:160)
  g1[6] = 0;
  g1[7] = 0;
  int32x4 z4 = {0, 0, 0, 0};
  int32x8 z8 = {0, 0, 0, 0, 0, 0, 0, 0};
  // 6-arg toolchain variant: (g0, g1, g2, g3, g4, cpol)
  __builtin_amdgcn_tensor_load_to_lds(g0, g1, z4, z4, z8, 0);
}

// ---------------------------------------------------------------------------
// Layout conversion kernels
// ---------------------------------------------------------------------------
__global__ void transpose_to_f16_kernel(const float* __restrict__ src,
                                        _Float16* __restrict__ dst,
                                        int K, int N) {
  int idx = blockIdx.x * blockDim.x + threadIdx.x;
  if (idx >= K * N) return;
  int n = idx / K;
  int k = idx - n * K;
  dst[(size_t)n * K + k] = (_Float16)src[(size_t)k * N + n];
}

__global__ void f32_to_f16_kernel(const float* __restrict__ src,
                                  _Float16* __restrict__ dst, int count) {
  int idx = blockIdx.x * blockDim.x + threadIdx.x;
  if (idx < count) dst[idx] = (_Float16)src[idx];
}

__global__ void row_copy_kernel(const float* __restrict__ src,
                                float* __restrict__ dst, int row, int N) {
  int idx = blockIdx.x * blockDim.x + threadIdx.x;
  if (idx < N) dst[idx] = src[(size_t)row * N + idx];
}

// ---------------------------------------------------------------------------
// Encoder: h = tanh(x0 @ enc_w1 + b1); [mean|logvar] = h @ enc_w2 + b2;
// z0 = mean + eps * exp(0.5*logvar).  4 WGs x 64 rows.
// ---------------------------------------------------------------------------
__global__ __launch_bounds__(256) void encoder_kernel(
    const _Float16* __restrict__ x0h,          // [256][512] f16
    const _Float16* __restrict__ w1t,          // [1024][512] f16
    const float* __restrict__ b1,
    const _Float16* __restrict__ w2t,          // [128][1024] f16
    const float* __restrict__ b2,
    const float* __restrict__ eps,
    float* __restrict__ z0ws,                  // [256][64] fp32
    _Float16* __restrict__ ztraj,              // t=0 slice
    float* __restrict__ out)
{
  __shared__ __align__(16) _Float16 hbuf[64 * ST64];
  __shared__ __align__(16) float    outbuf[64 * 128];

  const int tid  = threadIdx.x;
  const int wave = tid >> 5;
  const int lane = tid & 31;
  const int lr   = lane & 15;
  const int lhi  = lane >> 4;
  const int r0   = blockIdx.x * 64;

  floatx8 acc[4] = { f8zero(), f8zero(), f8zero(), f8zero() };

  for (int c2 = 0; c2 < ENC_H / 64; ++c2) {
    for (int j = 0; j < 2; ++j) {
      const int t  = 2 * wave + j;
      const int mi = t & 3;
      const int ni = t >> 2;
      floatx8 c = f8zero();
      for (int ks = 0; ks < OBS / 32; ++ks) {
        half16 a = load_frag(x0h + (size_t)(r0 + mi * 16) * OBS + ks * 32, OBS);
        half16 b = load_frag(w1t + (size_t)(c2 * 64 + ni * 16) * OBS + ks * 32, OBS);
        c = wmma_f16(a, b, c);
      }
      const int   hcol = c2 * 64 + ni * 16 + lr;
      const float bv   = b1[hcol];
      #pragma unroll
      for (int v = 0; v < 8; ++v)
        hbuf[(mi * 16 + v + 8 * lhi) * ST64 + ni * 16 + lr] =
            (_Float16)tanhf(c[v] + bv);
    }
    __syncthreads();
    for (int ks = 0; ks < 2; ++ks) {
      half16 bfrag = load_frag(
          w2t + (size_t)(wave * 16) * ENC_H + c2 * 64 + ks * 32, ENC_H);
      #pragma unroll
      for (int mt = 0; mt < 4; ++mt) {
        half16 afrag = load_frag(hbuf + mt * 16 * ST64 + ks * 32, ST64);
        acc[mt] = wmma_f16(afrag, bfrag, acc[mt]);
      }
    }
    __syncthreads();
  }

  {
    const int   n  = wave * 16 + lr;
    const float bv = b2[n];
    #pragma unroll
    for (int mt = 0; mt < 4; ++mt) {
      floatx8 c = acc[mt];
      #pragma unroll
      for (int v = 0; v < 8; ++v)
        outbuf[(mt * 16 + v + 8 * lhi) * 128 + n] = c[v] + bv;
    }
  }
  __syncthreads();

  const size_t MEAN_OFF = (size_t)TSTEPS * BATCH * OBS;
  const size_t LV_OFF   = MEAN_OFF + (size_t)BATCH * LAT;
  for (int i = tid; i < 64 * LAT; i += 256) {
    const int m    = i >> 6;
    const int n    = i & 63;
    const int row  = r0 + m;
    const float mean = outbuf[m * 128 + n];
    const float lv   = outbuf[m * 128 + 64 + n];
    const float z0v  = mean + eps[(size_t)row * LAT + n] * expf(0.5f * lv);
    out[MEAN_OFF + (size_t)row * LAT + n] = mean;
    out[LV_OFF   + (size_t)row * LAT + n] = lv;
    z0ws[(size_t)row * LAT + n]  = z0v;
    ztraj[(size_t)row * LAT + n] = (_Float16)z0v;
  }
}

// ---------------------------------------------------------------------------
// SDE Euler-Maruyama. 8 persistent WGs x 32 batch rows. Weights live in LDS
// (loaded once), fp32 z master state lives in registers (ownership map is
// step-invariant), f16 z copy in LDS for WMMA A-operands. concat([z,t]) is a
// K=64 GEMM + time-dependent bias b1 + t*W1[row 64].
// ---------------------------------------------------------------------------
__global__ __launch_bounds__(256) void sde_kernel(
    const float* __restrict__ ts, const float* __restrict__ dW,
    const _Float16* __restrict__ fw1t, const float* __restrict__ fw1l,
    const float* __restrict__ fb1,  const _Float16* __restrict__ fw2t,
    const float* __restrict__ fb2,
    const _Float16* __restrict__ gw1t, const float* __restrict__ gw1l,
    const float* __restrict__ gb1,  const _Float16* __restrict__ gw2t,
    const float* __restrict__ gb2,
    const float* __restrict__ z0, _Float16* __restrict__ ztraj)
{
  __shared__ __align__(16) _Float16 zh[32 * ST64];        // z f16
  __shared__ __align__(16) _Float16 hbs[32 * ST256];      // shared hidden
  __shared__ __align__(16) _Float16 lw1[2][SDE_H * ST64]; // f/g W1T [256][64]
  __shared__ __align__(16) _Float16 lw2[2][LAT * ST256];  // f/g W2T [64][256]

  const int tid  = threadIdx.x;
  const int wave = tid >> 5;
  const int lane = tid & 31;
  const int lr   = lane & 15;
  const int lhi  = lane >> 4;
  const int b0   = blockIdx.x * 32;

  // stage all four weight matrices into LDS (coalesced 16B copies)
  for (int i = tid; i < SDE_H * 8; i += 256) {       // W1: 256 rows x 64 halves
    const int row = i >> 3, seg = (i & 7) * 8;
    *(half8*)&lw1[0][row * ST64 + seg] = *(const half8*)&fw1t[row * 64 + seg];
    *(half8*)&lw1[1][row * ST64 + seg] = *(const half8*)&gw1t[row * 64 + seg];
  }
  for (int i = tid; i < LAT * 32; i += 256) {        // W2: 64 rows x 256 halves
    const int row = i >> 5, seg = (i & 31) * 8;
    *(half8*)&lw2[0][row * ST256 + seg] = *(const half8*)&fw2t[row * 256 + seg];
    *(half8*)&lw2[1][row * ST256 + seg] = *(const half8*)&gw2t[row * 256 + seg];
  }

  // per-thread fp32 z master state (GEMM2/update tile ownership)
  const int mi2 = wave & 1;
  const int ni2 = wave >> 1;
  const int n2  = ni2 * 16 + lr;
  floatx8 zreg;
  #pragma unroll
  for (int v = 0; v < 8; ++v) {
    const int m = mi2 * 16 + v + 8 * lhi;
    const float zv = z0[(size_t)(b0 + m) * LAT + n2];
    zreg[v] = zv;
    zh[m * ST64 + n2] = (_Float16)zv;
  }
  __syncthreads();

  for (int step = 0; step < TSTEPS - 1; ++step) {
    const float t   = ts[step];
    const float dt  = ts[step + 1] - t;
    const float sdt = sqrtf(dt);

    // hoisted A fragments of z
    half16 az[2][2];
    #pragma unroll
    for (int mi = 0; mi < 2; ++mi)
      #pragma unroll
      for (int ks = 0; ks < 2; ++ks)
        az[mi][ks] = load_frag(zh + mi * 16 * ST64 + ks * 32, ST64);

    floatx8 cd[2];
    for (int g = 0; g < 2; ++g) {
      const _Float16* w1l = &lw1[g][0];
      const float* bb1 = g ? gb1 : fb1;
      const float* bwl = g ? gw1l : fw1l;
      // layer 1: h = tanh(z @ W1 + (b1 + t*W1[64,:]))
      for (int j = 0; j < 2; ++j) {
        const int   ni  = 2 * wave + j;
        const int   col = ni * 16 + lr;
        const float bv  = bb1[col] + t * bwl[col];
        for (int mi = 0; mi < 2; ++mi) {
          floatx8 c = f8zero();
          #pragma unroll
          for (int ks = 0; ks < 2; ++ks)
            c = wmma_f16(az[mi][ks],
                         load_frag(w1l + (ni * 16) * ST64 + ks * 32, ST64), c);
          #pragma unroll
          for (int v = 0; v < 8; ++v)
            hbs[(mi * 16 + v + 8 * lhi) * ST256 + col] =
                (_Float16)tanhf(c[v] + bv);
        }
      }
      __syncthreads();                   // hidden ready
      // layer 2: one 16x16 tile per wave
      floatx8 c = f8zero();
      const _Float16* w2l = &lw2[g][0];
      for (int ks = 0; ks < SDE_H / 32; ++ks) {
        half16 a = load_frag(hbs + mi2 * 16 * ST256 + ks * 32, ST256);
        half16 b = load_frag(w2l + (ni2 * 16) * ST256 + ks * 32, ST256);
        c = wmma_f16(a, b, c);
      }
      cd[g] = c;
      __syncthreads();                   // hidden buffer reusable
    }

    const float fbv = fb2[n2];
    const float gbv = gb2[n2];
    #pragma unroll
    for (int v = 0; v < 8; ++v) {
      const int   m     = mi2 * 16 + v + 8 * lhi;
      const int   b     = b0 + m;
      const float drift = cd[0][v] + fbv;
      const float diff  = cd[1][v] + gbv;
      const float dw    = dW[((size_t)step * BATCH + b) * LAT + n2];
      const float zn    = zreg[v] + drift * dt + diff * sdt * dw;
      zreg[v] = zn;
      zh[m * ST64 + n2] = (_Float16)zn;
      ztraj[((size_t)(step + 1) * BATCH + b) * LAT + n2] = (_Float16)zn;
    }
    __syncthreads();                     // zh writes vs next-step az reads
  }
}

// ---------------------------------------------------------------------------
// Decoder (compute-dominant): x = tanh(z @ dec_w1 + b1) @ dec_w2 + b2.
// 800 WGs x 64-row tiles. W2 k-slabs (64KB) are DMA'd by the Tensor Data
// Mover into double-buffered LDS (padded rows), overlapping both GEMM stages.
// ---------------------------------------------------------------------------
__global__ __launch_bounds__(256) void decoder_kernel(
    const _Float16* __restrict__ ztraj,   // [T*256][64] f16
    const _Float16* __restrict__ w1t,     // [1024][64] f16
    const float* __restrict__ b1,
    const _Float16* __restrict__ w2t,     // [512][1024] f16
    const float* __restrict__ b2,
    float* __restrict__ out)
{
  __shared__ __align__(16) _Float16 hbuf[64 * ST64];
  __shared__ __align__(16) _Float16 wbuf[2][OBS * ST64]; // 2 x 72KB W2 slabs

  const int tid  = threadIdx.x;
  const int wave = tid >> 5;
  const int lane = tid & 31;
  const int lr   = lane & 15;
  const int lhi  = lane >> 4;
  const int r0   = blockIdx.x * 64;

  floatx8 acc[16];
  #pragma unroll
  for (int i = 0; i < 16; ++i) acc[i] = f8zero();

  // kick TDM for chunk 0
  if (wave == 0) {
    tdm_load_w2_chunk((unsigned)(uintptr_t)&wbuf[0][0], w2t);
    __builtin_amdgcn_s_wait_tensorcnt(0);
  }
  __syncthreads();

  // hoisted z A-fragments (K = LAT = 64)
  half16 az[2][2];
  #pragma unroll
  for (int j = 0; j < 2; ++j) {
    const int mi = (2 * wave + j) & 3;
    #pragma unroll
    for (int ks = 0; ks < 2; ++ks)
      az[j][ks] = load_frag(ztraj + (size_t)(r0 + mi * 16) * LAT + ks * 32, LAT);
  }

  for (int c2 = 0; c2 < DEC_H / 64; ++c2) {
    // prefetch next W2 slab via TDM (overlaps both GEMM stages)
    if (wave == 0 && c2 + 1 < DEC_H / 64)
      tdm_load_w2_chunk((unsigned)(uintptr_t)&wbuf[(c2 + 1) & 1][0],
                        w2t + (size_t)(c2 + 1) * 64);

    // ---- layer 1: hidden chunk 64x64 = tanh(z @ W1chunk + b1) ----
    for (int j = 0; j < 2; ++j) {
      const int t  = 2 * wave + j;
      const int mi = t & 3;
      const int ni = t >> 2;
      floatx8 c = f8zero();
      #pragma unroll
      for (int ks = 0; ks < 2; ++ks)
        c = wmma_f16(az[j][ks],
                     load_frag(w1t + (size_t)(c2 * 64 + ni * 16) * LAT + ks * 32,
                               LAT),
                     c);
      const int   hcol = c2 * 64 + ni * 16 + lr;
      const float bv   = b1[hcol];
      #pragma unroll
      for (int v = 0; v < 8; ++v)
        hbuf[(mi * 16 + v + 8 * lhi) * ST64 + ni * 16 + lr] =
            (_Float16)tanhf(c[v] + bv);
    }
    __syncthreads();

    // ---- layer 2 partial: out[64 x 512] += h_chunk @ W2slab (from LDS) ----
    const _Float16* wchunk = &wbuf[c2 & 1][0];
    for (int ks = 0; ks < 2; ++ks) {
      half16 af[4];
      #pragma unroll
      for (int mt = 0; mt < 4; ++mt)
        af[mt] = load_frag(hbuf + mt * 16 * ST64 + ks * 32, ST64);
      #pragma unroll
      for (int nt = 0; nt < 4; ++nt) {
        half16 bf = load_frag(wchunk + (wave * 64 + nt * 16) * ST64 + ks * 32,
                              ST64);
        #pragma unroll
        for (int mt = 0; mt < 4; ++mt)
          acc[mt * 4 + nt] = wmma_f16(af[mt], bf, acc[mt * 4 + nt]);
      }
    }
    if (wave == 0) __builtin_amdgcn_s_wait_tensorcnt(0);  // next slab landed
    __syncthreads();
  }

  #pragma unroll
  for (int nt = 0; nt < 4; ++nt) {
    const int   n  = wave * 64 + nt * 16 + lr;
    const float bv = b2[n];
    #pragma unroll
    for (int mt = 0; mt < 4; ++mt) {
      floatx8 c = acc[mt * 4 + nt];
      #pragma unroll
      for (int v = 0; v < 8; ++v)
        out[(size_t)(r0 + mt * 16 + v + 8 * lhi) * OBS + n] = c[v] + bv;
    }
  }
}

// ---------------------------------------------------------------------------
// Host launcher
// ---------------------------------------------------------------------------
extern "C" void kernel_launch(void* const* d_in, const int* in_sizes, int n_in,
                              void* d_out, int out_size, void* d_ws,
                              size_t ws_size, hipStream_t stream) {
  (void)in_sizes; (void)n_in; (void)out_size; (void)ws_size;

  const float* x0     = (const float*)d_in[0];
  const float* ts     = (const float*)d_in[1];
  const float* eps    = (const float*)d_in[2];
  const float* dW     = (const float*)d_in[3];
  const float* enc_w1 = (const float*)d_in[4];
  const float* enc_b1 = (const float*)d_in[5];
  const float* enc_w2 = (const float*)d_in[6];
  const float* enc_b2 = (const float*)d_in[7];
  const float* dec_w1 = (const float*)d_in[8];
  const float* dec_b1 = (const float*)d_in[9];
  const float* dec_w2 = (const float*)d_in[10];
  const float* dec_b2 = (const float*)d_in[11];
  const float* f_w1   = (const float*)d_in[12];
  const float* f_b1   = (const float*)d_in[13];
  const float* f_w2   = (const float*)d_in[14];
  const float* f_b2   = (const float*)d_in[15];
  const float* g_w1   = (const float*)d_in[16];
  const float* g_b1   = (const float*)d_in[17];
  const float* g_w2   = (const float*)d_in[18];
  const float* g_b2   = (const float*)d_in[19];
  float* out = (float*)d_out;

  char*  ws  = (char*)d_ws;
  size_t off = 0;
  auto take = [&](size_t bytes) -> char* {
    char* p = ws + off;
    off += (bytes + 255) & ~(size_t)255;
    return p;
  };

  _Float16* ew1t = (_Float16*)take((size_t)ENC_H * OBS * 2);
  _Float16* ew2t = (_Float16*)take((size_t)128 * ENC_H * 2);
  _Float16* dw1t = (_Float16*)take((size_t)DEC_H * LAT * 2);
  _Float16* dw2t = (_Float16*)take((size_t)OBS * DEC_H * 2);
  _Float16* fw1t = (_Float16*)take((size_t)SDE_H * LAT * 2);
  float*    fw1l = (float*)take((size_t)SDE_H * 4);
  _Float16* fw2t = (_Float16*)take((size_t)LAT * SDE_H * 2);
  _Float16* gw1t = (_Float16*)take((size_t)SDE_H * LAT * 2);
  float*    gw1l = (float*)take((size_t)SDE_H * 4);
  _Float16* gw2t = (_Float16*)take((size_t)LAT * SDE_H * 2);
  _Float16* x0h  = (_Float16*)take((size_t)BATCH * OBS * 2);
  float*    z0ws = (float*)take((size_t)BATCH * LAT * 4);
  _Float16* ztrj = (_Float16*)take((size_t)TSTEPS * BATCH * LAT * 2);

  auto cdiv = [](int a, int b) { return (a + b - 1) / b; };

  transpose_to_f16_kernel<<<cdiv(OBS * ENC_H, 256), 256, 0, stream>>>(
      enc_w1, ew1t, OBS, ENC_H);
  transpose_to_f16_kernel<<<cdiv(ENC_H * 128, 256), 256, 0, stream>>>(
      enc_w2, ew2t, ENC_H, 128);
  transpose_to_f16_kernel<<<cdiv(LAT * DEC_H, 256), 256, 0, stream>>>(
      dec_w1, dw1t, LAT, DEC_H);
  transpose_to_f16_kernel<<<cdiv(DEC_H * OBS, 256), 256, 0, stream>>>(
      dec_w2, dw2t, DEC_H, OBS);
  transpose_to_f16_kernel<<<cdiv(LAT * SDE_H, 256), 256, 0, stream>>>(
      f_w1, fw1t, LAT, SDE_H);
  transpose_to_f16_kernel<<<cdiv(SDE_H * LAT, 256), 256, 0, stream>>>(
      f_w2, fw2t, SDE_H, LAT);
  transpose_to_f16_kernel<<<cdiv(LAT * SDE_H, 256), 256, 0, stream>>>(
      g_w1, gw1t, LAT, SDE_H);
  transpose_to_f16_kernel<<<cdiv(SDE_H * LAT, 256), 256, 0, stream>>>(
      g_w2, gw2t, SDE_H, LAT);
  row_copy_kernel<<<1, 256, 0, stream>>>(f_w1, fw1l, LAT, SDE_H);
  row_copy_kernel<<<1, 256, 0, stream>>>(g_w1, gw1l, LAT, SDE_H);
  f32_to_f16_kernel<<<cdiv(BATCH * OBS, 256), 256, 0, stream>>>(
      x0, x0h, BATCH * OBS);

  encoder_kernel<<<BATCH / 64, 256, 0, stream>>>(
      x0h, ew1t, enc_b1, ew2t, enc_b2, eps, z0ws, ztrj, out);
  sde_kernel<<<BATCH / 32, 256, 0, stream>>>(
      ts, dW, fw1t, fw1l, f_b1, fw2t, f_b2,
      gw1t, gw1l, g_b1, gw2t, g_b2, z0ws, ztrj);
  decoder_kernel<<<(TSTEPS * BATCH) / 64, 256, 0, stream>>>(
      ztrj, dw1t, dec_b1, dw2t, dec_b2, out);
}
